// OAQLinear_24592982737457
// MI455X (gfx1250) — matrix operand order repaired and checked by reading, not compiled, not run on hardware
//
#include <hip/hip_runtime.h>
#include <hip/hip_bf16.h>

#define K_DIM 4096
#define N_DIM 11008
#define GROUPS (K_DIM / 128)

#define BM 128
#define BN 128
#define BK 32
#define LSTR 40  // halfs per LDS row: 32 data + 8 pad -> 80B row stride, conflict-free b128 frags

typedef __attribute__((ext_vector_type(16))) _Float16 v16h;
typedef __attribute__((ext_vector_type(8)))  _Float16 v8h;
typedef __attribute__((ext_vector_type(4)))  _Float16 v4h;
typedef __attribute__((ext_vector_type(8)))  float    v8f;
typedef __attribute__((ext_vector_type(4)))  float    v4f;
typedef __attribute__((ext_vector_type(4)))  int      v4i;

#if __has_builtin(__builtin_amdgcn_global_load_async_to_lds_b128) && \
    __has_builtin(__builtin_amdgcn_s_wait_asynccnt)
#define USE_ASYNC_LDS 1
#endif

// ---------------------------------------------------------------------------
// Kernel 1: dequantize w[n][k] = half( q[n][k] * scale[n][k/128] + outlier[n][k] )
// Pure streaming pass: ~450 MB of traffic, one-time cost so the GEMM reads fp16.
// ---------------------------------------------------------------------------
__global__ __launch_bounds__(256)
void oaq_dequant(const int* __restrict__ q, const float* __restrict__ s,
                 const float* __restrict__ ow, _Float16* __restrict__ w) {
  size_t idx = ((size_t)blockIdx.x * 256 + threadIdx.x) * 4;   // element index
  int row = (int)(idx >> 12);            // / 4096
  int col = (int)(idx & (K_DIM - 1));    // % 4096 (4-vector never crosses a 128-group)
  float scale = s[row * GROUPS + (col >> 7)];
  v4i qv = *(const v4i*)(q + idx);
  v4f ov = *(const v4f*)(ow + idx);
  v4h r;
#pragma unroll
  for (int i = 0; i < 4; ++i)
    r[i] = (_Float16)((float)qv[i] * scale + ov[i]);
  *(v4h*)(w + idx) = r;
}

// ---------------------------------------------------------------------------
// Fragment loaders from LDS (wave32 WMMA 16x16x32 f16 layouts per CDNA5 ISA).
// A (16x32, MxK): lane row = lane%16; halfs[0..7]=K[g*8..g*8+7], halfs[8..15]=K[16+g*8..]
// B (32x16, KxN) stored as W rows [n][k]: lane col = lane%16; halfs[0..7]=K[g*8..g*8+7], halfs[8..15]=K[16+g*8..]
// ---------------------------------------------------------------------------
__device__ __forceinline__ v16h frag_a(const _Float16* base, int lane) {
  const int row = lane & 15, g = lane >> 4;
  const _Float16* p = base + row * LSTR;
  union { v16h f; v8h h[2]; } u;
  u.h[0] = *(const v8h*)(p + g * 8);
  u.h[1] = *(const v8h*)(p + 16 + g * 8);
  return u.f;
}

__device__ __forceinline__ v16h frag_b(const _Float16* base, int lane) {
  const int col = lane & 15, g = lane >> 4;
  const _Float16* p = base + col * LSTR;
  union { v16h f; v8h h[2]; } u;
  u.h[0] = *(const v8h*)(p + g * 8);
  u.h[1] = *(const v8h*)(p + 16 + g * 8);
  return u.f;
}

#ifdef USE_ASYNC_LDS
// Async DMA: global (AS1) -> LDS (AS3), tracked by ASYNCcnt. 16 bytes per lane.
// Builtin params are typed: (int4 AS1*, int4 AS3*, imm offset, imm cpol).
__device__ __forceinline__ void async_ld16(const _Float16* gsrc, _Float16* ldst) {
  __builtin_amdgcn_global_load_async_to_lds_b128(
      (__attribute__((address_space(1))) v4i*)gsrc,
      (__attribute__((address_space(3))) v4i*)ldst, 0, 0);
}
#endif

// ---------------------------------------------------------------------------
// Kernel 2: C[M][N] = X[M][K] * W[N][K]^T + bias, fp32 accumulate, fp16 out.
// 256 threads = 8 waves (2 along M x 4 along N); wave tile 64x32 = 8 WMMA accs.
// LDS double buffered; global->LDS via async DMA (ASYNCcnt) when available,
// register-staged global prefetch otherwise. One barrier per K-step.
// ---------------------------------------------------------------------------
__global__ __launch_bounds__(256)
void oaq_gemm(const _Float16* __restrict__ X, const _Float16* __restrict__ W,
              const _Float16* __restrict__ bias, _Float16* __restrict__ Y) {
  __shared__ _Float16 lds[2][2][BM * LSTR];   // [stage][A|B] : 40 KB total

  const int t    = threadIdx.x;
  const int lane = t & 31;
  const int wid  = t >> 5;
  const int wm   = wid & 1;    // 0..1 -> which 64-row strip
  const int wn   = wid >> 1;   // 0..3 -> which 32-col strip
  const int bm   = blockIdx.y * BM;
  const int bn   = blockIdx.x * BN;

  // Cooperative tile load mapping: 128 rows x 32 halfs, 16B segments.
  // Thread t covers rows (t>>2) and (t>>2)+64, segment (t&3)*8 halfs.
  const int rowa = t >> 2;
  const int seg  = (t & 3) * 8;
  const _Float16* gA = X + (size_t)(bm + rowa) * K_DIM + seg;
  const _Float16* gB = W + (size_t)(bn + rowa) * K_DIM + seg;
  const size_t rstep = (size_t)64 * K_DIM;

  v8f acc[4][2] = {};

#ifdef USE_ASYNC_LDS
  // ---- prologue: DMA stage 0 ----
  {
    _Float16* lA = &lds[0][0][0];
    _Float16* lB = &lds[0][1][0];
    async_ld16(gA,         lA + rowa * LSTR + seg);
    async_ld16(gA + rstep, lA + (rowa + 64) * LSTR + seg);
    async_ld16(gB,         lB + rowa * LSTR + seg);
    async_ld16(gB + rstep, lB + (rowa + 64) * LSTR + seg);
  }
  __builtin_amdgcn_s_wait_asynccnt(0);
  __syncthreads();
#else
  v8h st0 = *(const v8h*)(gA);
  v8h st1 = *(const v8h*)(gA + rstep);
  v8h st2 = *(const v8h*)(gB);
  v8h st3 = *(const v8h*)(gB + rstep);
  {
    _Float16* lA = &lds[0][0][0];
    _Float16* lB = &lds[0][1][0];
    *(v8h*)(lA + rowa * LSTR + seg)        = st0;
    *(v8h*)(lA + (rowa + 64) * LSTR + seg) = st1;
    *(v8h*)(lB + rowa * LSTR + seg)        = st2;
    *(v8h*)(lB + (rowa + 64) * LSTR + seg) = st3;
  }
  __syncthreads();
#endif

#pragma unroll 1
  for (int kt = 0; kt < K_DIM / BK; ++kt) {
    const int  cur      = kt & 1;
    const bool has_next = (kt + 1) < (K_DIM / BK);

    if (has_next) {  // start next K-slab copy (overlaps with compute below)
      const _Float16* pa = gA + (size_t)(kt + 1) * BK;
      const _Float16* pb = gB + (size_t)(kt + 1) * BK;
#ifdef USE_ASYNC_LDS
      _Float16* nA = &lds[cur ^ 1][0][0];
      _Float16* nB = &lds[cur ^ 1][1][0];
      async_ld16(pa,         nA + rowa * LSTR + seg);
      async_ld16(pa + rstep, nA + (rowa + 64) * LSTR + seg);
      async_ld16(pb,         nB + rowa * LSTR + seg);
      async_ld16(pb + rstep, nB + (rowa + 64) * LSTR + seg);
#else
      st0 = *(const v8h*)(pa);
      st1 = *(const v8h*)(pa + rstep);
      st2 = *(const v8h*)(pb);
      st3 = *(const v8h*)(pb + rstep);
#endif
    }

    // ---- compute from current stage: preload all fragments, then 8 WMMAs ----
    const _Float16* lA = &lds[cur][0][0] + wm * 64 * LSTR;
    const _Float16* lB = &lds[cur][1][0] + wn * 32 * LSTR;
    v16h fb0 = frag_b(lB, lane);
    v16h fb1 = frag_b(lB + 16 * LSTR, lane);
    v16h fa0 = frag_a(lA + 0 * 16 * LSTR, lane);
    v16h fa1 = frag_a(lA + 1 * 16 * LSTR, lane);
    v16h fa2 = frag_a(lA + 2 * 16 * LSTR, lane);
    v16h fa3 = frag_a(lA + 3 * 16 * LSTR, lane);

    acc[0][0] = __builtin_amdgcn_wmma_f32_16x16x32_f16(false, fa0, false, fb0, (short)0, acc[0][0], false, false);
    acc[0][1] = __builtin_amdgcn_wmma_f32_16x16x32_f16(false, fa0, false, fb1, (short)0, acc[0][1], false, false);
    acc[1][0] = __builtin_amdgcn_wmma_f32_16x16x32_f16(false, fa1, false, fb0, (short)0, acc[1][0], false, false);
    acc[1][1] = __builtin_amdgcn_wmma_f32_16x16x32_f16(false, fa1, false, fb1, (short)0, acc[1][1], false, false);
    acc[2][0] = __builtin_amdgcn_wmma_f32_16x16x32_f16(false, fa2, false, fb0, (short)0, acc[2][0], false, false);
    acc[2][1] = __builtin_amdgcn_wmma_f32_16x16x32_f16(false, fa2, false, fb1, (short)0, acc[2][1], false, false);
    acc[3][0] = __builtin_amdgcn_wmma_f32_16x16x32_f16(false, fa3, false, fb0, (short)0, acc[3][0], false, false);
    acc[3][1] = __builtin_amdgcn_wmma_f32_16x16x32_f16(false, fa3, false, fb1, (short)0, acc[3][1], false, false);

    if (has_next) {
#ifdef USE_ASYNC_LDS
      __builtin_amdgcn_s_wait_asynccnt(0);  // next-stage DMA landed in LDS
      __syncthreads();
#else
      _Float16* nA = &lds[cur ^ 1][0][0];
      _Float16* nB = &lds[cur ^ 1][1][0];
      *(v8h*)(nA + rowa * LSTR + seg)        = st0;
      *(v8h*)(nA + (rowa + 64) * LSTR + seg) = st1;
      *(v8h*)(nB + rowa * LSTR + seg)        = st2;
      *(v8h*)(nB + (rowa + 64) * LSTR + seg) = st3;
      __syncthreads();
#endif
    }
  }

  // ---- epilogue: bias + fp16 store per C/D VGPR layout ----
  const int gcol = lane & 15;
  const int ghi  = (lane >> 4) << 3;   // lanes 16..31 hold rows m+8..m+15
#pragma unroll
  for (int j = 0; j < 2; ++j) {
    const int n = bn + wn * 32 + j * 16 + gcol;
    const float bv = (float)bias[n];
#pragma unroll
    for (int i = 0; i < 4; ++i) {
      const int m0 = bm + wm * 64 + i * 16 + ghi;
#pragma unroll
      for (int r = 0; r < 8; ++r) {
        Y[(size_t)(m0 + r) * N_DIM + n] = (_Float16)(acc[i][j][r] + bv);
      }
    }
  }
}

// ---------------------------------------------------------------------------
extern "C" void kernel_launch(void* const* d_in, const int* in_sizes, int n_in,
                              void* d_out, int out_size, void* d_ws, size_t ws_size,
                              hipStream_t stream) {
  const _Float16* x      = (const _Float16*)d_in[0];  // [4,2048,4096] fp16
  const int*      qw     = (const int*)d_in[1];       // [11008,4096] int32 (4-bit codes)
  const float*    scales = (const float*)d_in[2];     // [11008,32] fp32
  const float*    outw   = (const float*)d_in[3];     // [11008,4096] fp32
  const _Float16* bias   = (const _Float16*)d_in[4];  // [11008] fp16
  _Float16* y    = (_Float16*)d_out;                  // [4,2048,11008] fp16
  _Float16* wdeq = (_Float16*)d_ws;                   // 11008*4096*2 = 90.2 MB scratch

  const int M = in_sizes[0] / K_DIM;  // 8192

  // 1) dequantize weights to fp16 once (memory-bound streaming pass)
  const int total_vec = (N_DIM * K_DIM) / 4;
  oaq_dequant<<<total_vec / 256, 256, 0, stream>>>(qw, scales, outw, wdeq);

  // 2) compute-bound fp16 WMMA GEMM with fused bias
  dim3 grid(N_DIM / BN, M / BM);
  oaq_gemm<<<grid, 256, 0, stream>>>(x, wdeq, bias, y);
}
